// InfoNCELoss_74474732913209
// MI455X (gfx1250) — matrix operand order repaired and checked by reading, not compiled, not run on hardware
//
#include <hip/hip_runtime.h>
#include <hip/hip_bf16.h>

// Problem constants (fixed by setup_inputs in the reference)
#define TOTAL   4096        // rows of embeddings (kN)
#define DIM     512         // embedding dim
#define NPAIR   2048        // number of positive pairs (num_augs = 2)
#define INV_T   14.2857142857142857f  // 1 / 0.07

#define BT          128     // macro-tile edge (rows and cols per block)
#define KSTEP       32      // K elements per pipeline stage (one bf16 WMMA K)
#define NSTEP       (DIM / KSTEP)   // 16 stages
#define LDS_ROW_B   80      // padded LDS row: 64 B data + 16 B pad (bank-conflict-free)

typedef __bf16 v4bf  __attribute__((ext_vector_type(4)));
typedef __bf16 v8bf  __attribute__((ext_vector_type(8)));
typedef __bf16 v16bf __attribute__((ext_vector_type(16)));
typedef float  v8f   __attribute__((ext_vector_type(8)));
typedef int    i4v   __attribute__((vector_size(16)));   // matches builtin proto

// ---- gfx1250 async global->LDS path (ASYNCcnt), with compile-safe fallback --
#if defined(__has_builtin)
#if __has_builtin(__builtin_amdgcn_global_load_async_to_lds_b128) && \
    __has_builtin(__builtin_amdgcn_s_wait_asynccnt)
#define HAVE_ASYNC 1
#endif
#endif

__device__ __forceinline__ void copy16_async(const void* g, void* l) {
#ifdef HAVE_ASYNC
  __builtin_amdgcn_global_load_async_to_lds_b128(
      (__attribute__((address_space(1))) i4v*)g,
      (__attribute__((address_space(3))) i4v*)l, 0, 0);
#else
  *(uint4*)l = *(const uint4*)g;      // VGPR-staged fallback
#endif
}

#ifdef HAVE_ASYNC
#define WAIT_ASYNC(n) __builtin_amdgcn_s_wait_asynccnt(n)
#else
#define WAIT_ASYNC(n)
#endif

// ---------------------------------------------------------------------------
// Pass 0: L2-normalize rows (fp32 -> bf16 in workspace); zero neg accumulator.
// ---------------------------------------------------------------------------
__global__ void __launch_bounds__(128)
normalize_kernel(const float* __restrict__ emb,
                 __bf16* __restrict__ nbf,
                 float* __restrict__ neg_accum) {
  __shared__ float red[128];
  const int row = blockIdx.x;
  const int tid = threadIdx.x;

  const float4 v = ((const float4*)(emb + (size_t)row * DIM))[tid];
  red[tid] = v.x * v.x + v.y * v.y + v.z * v.z + v.w * v.w;
  __syncthreads();
  for (int off = 64; off > 0; off >>= 1) {
    if (tid < off) red[tid] += red[tid + off];
    __syncthreads();
  }
  const float inv = rsqrtf(red[0]);

  v4bf o;
  o[0] = (__bf16)(v.x * inv);
  o[1] = (__bf16)(v.y * inv);
  o[2] = (__bf16)(v.z * inv);
  o[3] = (__bf16)(v.w * inv);
  ((v4bf*)(nbf + (size_t)row * DIM))[tid] = o;

  if (row == 0 && tid == 0) *neg_accum = 0.0f;
}

// ---------------------------------------------------------------------------
// Pass 1: LDS-tiled, double-buffered, async-fed WMMA similarity GEMM fused
// with the masked exp reduction.
//   block (8 waves) -> 128x128 tile of S; wave w -> 16-row strip x 128 cols
//   (8 v8f accumulators). Per stage: async-copy 128x32 A and B panels into
//   LDS (padded rows), s_wait_asynccnt + barrier, 8 WMMAs per wave from LDS.
// Only blocks with tile-col >= tile-row run; diagonal blocks mask i<j.
// ---------------------------------------------------------------------------
__global__ void __launch_bounds__(256)
simgemm_kernel(const __bf16* __restrict__ nbf,
               const int* __restrict__ labels,
               float* __restrict__ posS,
               float* __restrict__ neg_accum) {
  const int tib = blockIdx.y;
  const int tjb = blockIdx.x;
  if (tjb < tib) return;                 // block-uniform: nobody reaches barriers

  __shared__ __align__(16) char lds[2][2][BT * LDS_ROW_B];   // [buf][A|B][...]

  const int tid  = threadIdx.x;
  const int wave = tid >> 5;
  const int lane = tid & 31;
  const int half = lane >> 4;            // 0 | 1
  const int l15  = lane & 15;
  const int I0 = tib * BT, J0 = tjb * BT;

  // Copy role: threads 0..127 feed the A panel, 128..255 the B panel;
  // each thread owns one 64 B row segment (4 async b128 ops per stage).
  const int   crow   = tid & 127;
  const int   cpanel = tid >> 7;
  const char* gsrc0  = (const char*)nbf +
                       ((size_t)((cpanel ? J0 : I0) + crow)) * (DIM * 2);
  char* lrow[2] = { &lds[0][cpanel][crow * LDS_ROW_B],
                    &lds[1][cpanel][crow * LDS_ROW_B] };

  v8f acc[8] = {};

  // Prologue: stage 0 into buffer 0
#pragma unroll
  for (int p = 0; p < 4; ++p) copy16_async(gsrc0 + p * 16, lrow[0] + p * 16);

  for (int step = 0; step < NSTEP; ++step) {
    const int buf = step & 1;
    if (step + 1 < NSTEP) {
      const char* g = gsrc0 + (size_t)(step + 1) * (KSTEP * 2);
#pragma unroll
      for (int p = 0; p < 4; ++p) copy16_async(g + p * 16, lrow[buf ^ 1] + p * 16);
      WAIT_ASYNC(4);                     // in-order: current stage's 4 done
    } else {
      WAIT_ASYNC(0);
    }
    __syncthreads();                     // panels visible to all waves

    // A fragment (16x32 bf16): lane -> row l15, K = half*8+[0,8) and +16
    const char* Ap = &lds[buf][0][(wave * 16 + l15) * LDS_ROW_B + half * 16];
    v8bf a0 = *(const v8bf*)(Ap);
    v8bf a1 = *(const v8bf*)(Ap + 32);
    v16bf a = __builtin_shufflevector(a0, a1,
              0,1,2,3,4,5,6,7,8,9,10,11,12,13,14,15);

#pragma unroll
    for (int n = 0; n < 8; ++n) {
      // B fragment (32x16): lane -> row n*16+l15, K = half*8+[0,8) and +16
      const char* Bp = &lds[buf][1][(n * 16 + l15) * LDS_ROW_B + half * 16];
      v8bf b0 = *(const v8bf*)(Bp);
      v8bf b1 = *(const v8bf*)(Bp + 32);
      v16bf b = __builtin_shufflevector(b0, b1,
                0,1,2,3,4,5,6,7,8,9,10,11,12,13,14,15);
      acc[n] = __builtin_amdgcn_wmma_f32_16x16x32_bf16(
          false, a, false, b, (short)0, acc[n], false, false);
    }
    __syncthreads();                     // all reads done before next overwrite
  }

  // ----- Epilogue: mask + exp reduce; record positive similarities --------
  const int ibase = I0 + wave * 16 + (half << 3);
  int li[8];
#pragma unroll
  for (int r = 0; r < 8; ++r) li[r] = labels[(ibase + r) >> 1];

  float neg = 0.0f;
#pragma unroll
  for (int n = 0; n < 8; ++n) {
    const int j  = J0 + n * 16 + l15;
    const int lj = labels[j >> 1];
#pragma unroll
    for (int r = 0; r < 8; ++r) {
      const int   i = ibase + r;
      const float s = acc[n][r] * INV_T;
      if (i < j && li[r] != lj) neg += __expf(s);              // negative pair
      if ((i + 1 == j) && ((i & 1) == 0)) posS[i >> 1] = s;    // positive pair
    }
  }
  for (int off = 16; off > 0; off >>= 1)
    neg += __shfl_xor(neg, off, 32);
  if (lane == 0) atomicAdd(neg_accum, neg);
}

// ---------------------------------------------------------------------------
// Pass 2: loss = mean_t( log(exp(S_p[t]) + neg_sum) - S_p[t] )
// ---------------------------------------------------------------------------
__global__ void __launch_bounds__(256)
finalize_kernel(const float* __restrict__ posS,
                const float* __restrict__ neg_accum,
                float* __restrict__ out) {
  __shared__ float red[256];
  const float neg = *neg_accum;
  float sum = 0.0f;
  for (int t = threadIdx.x; t < NPAIR; t += 256) {
    const float s = posS[t];
    sum += logf(expf(s) + neg) - s;
  }
  red[threadIdx.x] = sum;
  __syncthreads();
  for (int off = 128; off > 0; off >>= 1) {
    if (threadIdx.x < off) red[threadIdx.x] += red[threadIdx.x + off];
    __syncthreads();
  }
  if (threadIdx.x == 0) out[0] = red[0] * (1.0f / (float)NPAIR);
}

// ---------------------------------------------------------------------------
extern "C" void kernel_launch(void* const* d_in, const int* in_sizes, int n_in,
                              void* d_out, int out_size, void* d_ws, size_t ws_size,
                              hipStream_t stream) {
  const float* emb    = (const float*)d_in[0];   // [4096, 512] fp32
  const int*   labels = (const int*)d_in[1];     // [2048] int32
  float*       out    = (float*)d_out;           // scalar loss

  // Workspace: bf16 normalized matrix | posS[2048] | neg accumulator
  __bf16* nbf    = (__bf16*)d_ws;
  float*  posS   = (float*)((char*)d_ws + (size_t)TOTAL * DIM * sizeof(__bf16));
  float*  negacc = posS + NPAIR;

  normalize_kernel<<<TOTAL, 128, 0, stream>>>(emb, nbf, negacc);

  simgemm_kernel<<<dim3(TOTAL / BT, TOTAL / BT), 256, 0, stream>>>(
      nbf, labels, posS, negacc);

  finalize_kernel<<<1, 256, 0, stream>>>(posS, negacc, out);
}
